// Graph_Pertu_Strategy_91302414778879
// MI455X (gfx1250) — matrix operand order repaired and checked by reading, not compiled, not run on hardware
//
#include <hip/hip_runtime.h>
#include <hip/hip_bf16.h>
#include <math.h>

// ---------------------------------------------------------------------------
// Types
// ---------------------------------------------------------------------------
typedef __attribute__((ext_vector_type(16))) _Float16     v16h;
typedef __attribute__((ext_vector_type(8)))  float        v8f;
typedef __attribute__((ext_vector_type(4)))  unsigned int u32x4;
typedef __attribute__((ext_vector_type(8)))  int          i32x8;
typedef __attribute__((ext_vector_type(4)))  int          i32x4;

#define DF        256     // D_FEAT (fixed by reference)
#define TILE_ROWS 32      // rows per TDM tile (32 rows * 1KB = 32KB LDS)

#if defined(__HIP_DEVICE_COMPILE__)
  #if __has_builtin(__builtin_amdgcn_tensor_load_to_lds) && \
      __has_builtin(__builtin_amdgcn_tensor_store_from_lds)
    #define USE_TDM 1
  #else
    #define USE_TDM 0
  #endif
#else
  #define USE_TDM 0
#endif

// ---------------------------------------------------------------------------
// threefry2x32 (exact JAX PRNG core), usable host-side for key split
// ---------------------------------------------------------------------------
__host__ __device__ static inline void threefry2x32(unsigned k0, unsigned k1,
                                                    unsigned c0, unsigned c1,
                                                    unsigned& o0, unsigned& o1) {
  unsigned ks2 = k0 ^ k1 ^ 0x1BD11BDAu;
  unsigned x0 = c0 + k0, x1 = c1 + k1;
#define TFR(r) { x0 += x1; x1 = (x1 << (r)) | (x1 >> (32 - (r))); x1 ^= x0; }
  TFR(13) TFR(15) TFR(26) TFR(6)  x0 += k1;  x1 += ks2 + 1u;
  TFR(17) TFR(29) TFR(16) TFR(24) x0 += ks2; x1 += k0 + 2u;
  TFR(13) TFR(15) TFR(26) TFR(6)  x0 += k0;  x1 += k1 + 3u;
  TFR(17) TFR(29) TFR(16) TFR(24) x0 += k1;  x1 += ks2 + 4u;
  TFR(13) TFR(15) TFR(26) TFR(6)  x0 += ks2; x1 += k0 + 5u;
#undef TFR
  o0 = x0; o1 = x1;
}

__device__ static inline float u01_from_bits(unsigned b) {
  // JAX uniform: bitcast((bits >> 9) | 0x3f800000) - 1.0
  return __uint_as_float((b >> 9) | 0x3f800000u) - 1.0f;
}

// searchsorted(ptr, v, side='right') - 1  over ptr[0..ng]
__device__ static inline int egid_of(const int* __restrict__ ptr, int ng, int v) {
  int lo = 0, hi = ng + 1;
  while (lo < hi) { int mid = (lo + hi) >> 1; if (ptr[mid] <= v) lo = mid + 1; else hi = mid; }
  return lo - 1;
}

// ---------------------------------------------------------------------------
// Kernel 0: zero the node-drop bitmask in workspace
// ---------------------------------------------------------------------------
__global__ void init_bits_k(unsigned* __restrict__ bits, int nwords) {
  int i = blockIdx.x * blockDim.x + threadIdx.x;
  if (i < nwords) bits[i] = 0u;
}

// ---------------------------------------------------------------------------
// Kernel 1: tiny MLP via WMMA (1 wave).  ratios = sigmoid(relu(ad2@W1+b1)@W2+b2)
//   writes ws_drop[g] = drop_ratio, ws_dropnum[g] = floor(node_num*mask_ratio)
// Layouts (wave32, 16x16x32 f16 WMMA):
//   A (16x32): lane L -> M=L%16 ; elem e -> K = (e>=8)*16 + (L>=16)*8 + (e&7)
//   B (32x16): lane L -> N=L%16 ; elem e -> K = (e>=8)*16 + (L>=16)*8 + (e&7)
//   C/D(16x16 f32): vgpr r, lane L -> M = r + (L>=16)*8, N = L%16
// ---------------------------------------------------------------------------
__global__ __launch_bounds__(32) void mlp_wmma_k(
    const float* __restrict__ adeg, const int* __restrict__ ptr,
    const float* __restrict__ W1, const float* __restrict__ b1,
    const float* __restrict__ W2, const float* __restrict__ b2,
    int ng, float* __restrict__ ws_drop, int* __restrict__ ws_dropnum) {
  __shared__ float hbuf[16 * 32];
  const int lane = threadIdx.x;
  const int mrow = lane & 15;
  const int hi16 = (lane >> 4) & 1;
  const int mtiles = (ng + 15) / 16;

  for (int mt = 0; mt < mtiles; ++mt) {
    // ---- layer 1: h = relu(ad2 @ W1 + b1),  ad2[:,k]=avg_degree, k<2
    v16h a;
    for (int e = 0; e < 16; ++e) {
      int k = ((e >= 8) ? 16 : 0) + (hi16 ? 8 : 0) + (e & 7);
      int m = mt * 16 + mrow;
      float v = (k < 2 && m < ng) ? adeg[m] : 0.0f;
      a[e] = (_Float16)v;
    }
    for (int nt = 0; nt < 2; ++nt) {
      int n = nt * 16 + mrow;
      v16h bm;
      for (int e = 0; e < 16; ++e) {
        int k = ((e >= 8) ? 16 : 0) + (hi16 ? 8 : 0) + (e & 7);
        float v = (k < 2) ? W1[k * 32 + n] : 0.0f;
        bm[e] = (_Float16)v;
      }
      v8f c = {};
      c = __builtin_amdgcn_wmma_f32_16x16x32_f16(false, a, false, bm,
                                                 (short)0, c, false, false);
      for (int r = 0; r < 8; ++r) {
        int ml = r + (hi16 ? 8 : 0);
        float h = c[r] + b1[n];
        hbuf[ml * 32 + n] = (h > 0.0f) ? h : 0.0f;
      }
    }
    __syncthreads();

    // ---- layer 2: ratios = sigmoid(h @ W2 + b2), K=32 exact, N<2 used
    v16h a2;
    for (int e = 0; e < 16; ++e) {
      int k = ((e >= 8) ? 16 : 0) + (hi16 ? 8 : 0) + (e & 7);
      a2[e] = (_Float16)hbuf[mrow * 32 + k];
    }
    const int n2 = mrow;
    v16h b2m;
    for (int e = 0; e < 16; ++e) {
      int k = ((e >= 8) ? 16 : 0) + (hi16 ? 8 : 0) + (e & 7);
      float v = (n2 < 2) ? W2[k * 2 + n2] : 0.0f;
      b2m[e] = (_Float16)v;
    }
    v8f c2 = {};
    c2 = __builtin_amdgcn_wmma_f32_16x16x32_f16(false, a2, false, b2m,
                                                (short)0, c2, false, false);
    for (int r = 0; r < 8; ++r) {
      int ml = r + (hi16 ? 8 : 0);
      int m = mt * 16 + ml;
      if (m < ng && n2 < 2) {
        float s = 1.0f / (1.0f + expf(-(c2[r] + b2[n2])));
        if (n2 == 0) {
          ws_drop[m] = s;                       // drop_ratio
        } else {
          int nn = ptr[m + 1] - ptr[m];         // node_num
          ws_dropnum[m] = (int)floorf((float)nn * s);
        }
      }
    }
    __syncthreads();
  }
}

// ---------------------------------------------------------------------------
// Kernel 2: per-edge keep mask.  Thread t handles edges t and t+half
// (threefry counter pairing matches jax.random.uniform over iota(n_edges)).
// ---------------------------------------------------------------------------
__global__ void edge_keep_k(const int* __restrict__ edge_src,
                            const int* __restrict__ ptr,
                            const float* __restrict__ ws_drop,
                            int ng, int n_edges, unsigned ke0, unsigned ke1,
                            float* __restrict__ out_keep) {
  int t = blockIdx.x * blockDim.x + threadIdx.x;
  int half = n_edges >> 1;
  if (t >= half) return;
  unsigned o0, o1;
  threefry2x32(ke0, ke1, (unsigned)t, (unsigned)(t + half), o0, o1);
  for (int s = 0; s < 2; ++s) {
    int e = t + s * half;
    float u = u01_from_bits(s ? o1 : o0);
    int g = egid_of(ptr, ng, edge_src[e]);
    out_keep[e] = (u >= ws_drop[g]) ? 1.0f : 0.0f;
  }
}

// ---------------------------------------------------------------------------
// Kernel 3: per-graph random ranking -> node-drop bitmask.
// key = (float)g + u_n  (the reference's argsort key), stable ties by index.
// ---------------------------------------------------------------------------
__global__ void rank_k(const int* __restrict__ ptr,
                       const int* __restrict__ ws_dropnum, int n_nodes,
                       unsigned kn0, unsigned kn1, unsigned* __restrict__ bits) {
  __shared__ float keys[1024];
  const int g = blockIdx.x;
  const int beg = ptr[g];
  int n = ptr[g + 1] - beg;
  if (n > 1024) n = 1024;
  const int half = n_nodes >> 1;

  for (int li = threadIdx.x; li < n; li += blockDim.x) {
    int i = beg + li;
    unsigned c0, c1, o0, o1;
    if (i < half) { c0 = (unsigned)i; c1 = (unsigned)(i + half); }
    else          { c0 = (unsigned)(i - half); c1 = (unsigned)i; }
    threefry2x32(kn0, kn1, c0, c1, o0, o1);
    float u = u01_from_bits((i < half) ? o0 : o1);
    keys[li] = (float)g + u;
  }
  __syncthreads();

  const int dn = ws_dropnum[g];
  for (int li = threadIdx.x; li < n; li += blockDim.x) {
    float ki = keys[li];
    int rank = 0;
    for (int j = 0; j < n; ++j) {
      float kj = keys[j];
      rank += (kj < ki) || (kj == ki && j < li);
    }
    if (rank < dn) {
      int i = beg + li;
      atomicOr(&bits[i >> 5], 1u << (i & 31));
    }
  }
}

// ---------------------------------------------------------------------------
// Kernel 4: new_x masked streaming copy via Tensor Data Mover.
// One wave per WG: TDM load 32 rows -> LDS, zero dropped rows, TDM store.
// D# built per CDNA5 ISA §8.3/8.4 (count=1, type=2, data_size=4B, 1-D tile).
// ---------------------------------------------------------------------------
__device__ static inline void make_tdm_desc(unsigned long long ga,
                                            unsigned lds_addr, unsigned nelems,
                                            u32x4& g0, i32x8& g1) {
  g0.x = 1u;                                               // count=1, user desc
  g0.y = lds_addr;                                         // LDS byte address
  g0.z = (unsigned)ga;                                     // global_addr[31:0]
  g0.w = (unsigned)((ga >> 32) & 0x01FFFFFFull) | (2u << 30);  // addr[56:32] | type=2
  unsigned td0 = nelems, td1 = 1u, tile0 = nelems;         // 1-D tile
  g1[0] = (int)(2u << 16);                                 // data_size=2 (4 bytes)
  g1[1] = (int)((td0 & 0xFFFFu) << 16);                    // tensor_dim0 lo16
  g1[2] = (int)((td0 >> 16) | ((td1 & 0xFFFFu) << 16));    // dim0 hi | dim1 lo
  g1[3] = (int)((td1 >> 16) | (tile0 << 16));              // dim1 hi | tile_dim0
  g1[4] = 0;                                               // tile_dim1/2 unused
  g1[5] = (int)td0;                                        // dim0_stride lo32
  g1[6] = 0;                                               // stride hi / dim1_stride lo
  g1[7] = 0;
}

__global__ __launch_bounds__(32) void mask_copy_k(const float* __restrict__ x,
                                                  const unsigned* __restrict__ bits,
                                                  float* __restrict__ out,
                                                  int n_rows) {
  __shared__ __align__(16) float tile[TILE_ROWS * DF];
  const int blk = blockIdx.x;
  const int lane = threadIdx.x;
  const int row0 = blk * TILE_ROWS;
  int rows = n_rows - row0;
  if (rows <= 0) return;
  if (rows > TILE_ROWS) rows = TILE_ROWS;
  const int nelems = rows * DF;
  const unsigned w = bits[blk];   // TILE_ROWS==32 -> one bitmask word per block

#if USE_TDM
  {
    unsigned lds_addr = (unsigned)(unsigned long long)(void*)&tile[0];
    unsigned long long ga = (unsigned long long)(const void*)(x + (long long)row0 * DF);
    u32x4 g0; i32x8 g1;
    i32x4 gz4 = {0, 0, 0, 0};
    i32x8 gz8 = {0, 0, 0, 0, 0, 0, 0, 0};
    make_tdm_desc(ga, lds_addr, (unsigned)nelems, g0, g1);
    __builtin_amdgcn_tensor_load_to_lds(g0, g1, gz4, gz4, gz8, 0);
    __builtin_amdgcn_s_wait_tensorcnt(0);

    // zero-patch dropped rows in LDS (cooperative, 2 x float4 per lane per row)
    for (int r = 0; r < rows; ++r) {
      if ((w >> r) & 1u) {
        float4 z = make_float4(0.f, 0.f, 0.f, 0.f);
        float4* rowp = (float4*)&tile[r * DF];
        rowp[lane] = z;
        rowp[lane + 32] = z;
      }
    }
    __asm__ volatile("s_wait_dscnt 0" ::: "memory");  // LDS visible to TDM

    unsigned long long go = (unsigned long long)(void*)(out + (long long)row0 * DF);
    make_tdm_desc(go, lds_addr, (unsigned)nelems, g0, g1);
    __builtin_amdgcn_tensor_store_from_lds(g0, g1, gz4, gz4, gz8, 0);
    __builtin_amdgcn_s_wait_tensorcnt(0);
  }
#else
  {
    // Fallback: plain wide streaming copy (b128 per lane)
    const float4* xin = (const float4*)(x + (long long)row0 * DF);
    float4*       po  = (float4*)(out + (long long)row0 * DF);
    const int nv = nelems / 4;                 // 64 float4 per row
    for (int c = lane; c < nv; c += 32) {
      int r = c >> 6;
      float4 v = xin[c];
      if ((w >> r) & 1u) v = make_float4(0.f, 0.f, 0.f, 0.f);
      po[c] = v;
    }
  }
#endif
}

// ---------------------------------------------------------------------------
// Host launcher
// ---------------------------------------------------------------------------
extern "C" void kernel_launch(void* const* d_in, const int* in_sizes, int n_in,
                              void* d_out, int out_size, void* d_ws, size_t ws_size,
                              hipStream_t stream) {
  const float* x    = (const float*)d_in[0];
  const int*   eidx = (const int*)d_in[1];   // row 0 = source nodes
  const int*   ptr  = (const int*)d_in[2];
  const float* adeg = (const float*)d_in[3];
  const float* W1   = (const float*)d_in[4];
  const float* b1   = (const float*)d_in[5];
  const float* W2   = (const float*)d_in[6];
  const float* b2   = (const float*)d_in[7];

  const int n_edges = in_sizes[1] / 2;
  const int ng      = in_sizes[3];
  const int n_nodes = in_sizes[0] / DF;

  // workspace layout: drop_ratio[128] | dropnum[128] | drop-bitmask
  float*    ws_drop    = (float*)d_ws;
  int*      ws_dropnum = (int*)((char*)d_ws + 512);
  unsigned* ws_bits    = (unsigned*)((char*)d_ws + 1024);
  const int nwords     = (n_nodes + 31) / 32;

  // jax.random.split(jax.random.key(42)) on host: counts [0,1,2,3] -> pairs (0,2),(1,3)
  unsigned x02, y02, x13, y13;
  threefry2x32(0u, 42u, 0u, 2u, x02, y02);
  threefry2x32(0u, 42u, 1u, 3u, x13, y13);
  const unsigned ke0 = x02, ke1 = x13;   // edge key
  const unsigned kn0 = y02, kn1 = y13;   // node key

  float* out_keep = (float*)d_out;
  float* out_x    = out_keep + n_edges;

  init_bits_k<<<dim3((nwords + 255) / 256), dim3(256), 0, stream>>>(ws_bits, nwords);
  mlp_wmma_k<<<dim3(1), dim3(32), 0, stream>>>(adeg, ptr, W1, b1, W2, b2,
                                               ng, ws_drop, ws_dropnum);
  edge_keep_k<<<dim3((n_edges / 2 + 255) / 256), dim3(256), 0, stream>>>(
      eidx, ptr, ws_drop, ng, n_edges, ke0, ke1, out_keep);
  rank_k<<<dim3(ng), dim3(256), 0, stream>>>(ptr, ws_dropnum, n_nodes,
                                             kn0, kn1, ws_bits);
  mask_copy_k<<<dim3((n_nodes + TILE_ROWS - 1) / TILE_ROWS), dim3(32), 0, stream>>>(
      x, ws_bits, out_x, n_nodes);
}
